// TextEncoder_1503238554057
// MI455X (gfx1250) — compile-verified, matrix operand order checked by reading, not run on hardware
//
#include <hip/hip_runtime.h>
#include <cstddef>
#include <cstdint>

// ---------------- problem constants ----------------
constexpr int C   = 192;
constexpr int DFF = 768;
constexpr int NH  = 2;
constexpr int NL  = 6;
constexpr int KS  = 3;
constexpr int OUT = 192;
constexpr int B   = 64;
constexpr int T   = 512;
constexpr int RW  = 4;          // relative window
constexpr int DH  = C / NH;     // 96
constexpr int SC  = 3 * C;      // 576
constexpr int M   = B * T;      // 32768 rows
constexpr int NREL = (2 * RW + 1) * DH;      // 864 bf16 elems per table
constexpr int NRELU16 = NREL / 8;            // 108 16-byte units

// ---------------- types ----------------
typedef __bf16 bf16_t;
typedef __attribute__((ext_vector_type(16))) __bf16 v16bf;
typedef __attribute__((ext_vector_type(8)))  __bf16 v8bf;
typedef __attribute__((ext_vector_type(8)))  float  v8f;

__device__ __forceinline__ v8f wmma_bf16(v16bf a, v16bf b, v8f c) {
  return __builtin_amdgcn_wmma_f32_16x16x32_bf16(false, a, false, b, (short)0, c,
                                                 false, false);
}

// per-lane operand fragment: elements [0..7]=p[0..7], [8..15]=p[16..23]
__device__ __forceinline__ v16bf load_row16(const bf16_t* p) {
  v8bf lo = *reinterpret_cast<const v8bf*>(p);
  v8bf hi = *reinterpret_cast<const v8bf*>(p + 16);
  return __builtin_shufflevector(lo, hi, 0, 1, 2, 3, 4, 5, 6, 7,
                                 8, 9, 10, 11, 12, 13, 14, 15);
}

// async 16-byte global -> LDS copy (gfx1250, tracked by ASYNCcnt)
__device__ __forceinline__ void async_copy_b128(bf16_t* lds_dst, const bf16_t* gsrc) {
  uint32_t lds_off = (uint32_t)(uintptr_t)lds_dst;
  asm volatile("global_load_async_to_lds_b128 %0, %1, off"
               :: "v"(lds_off), "v"(gsrc) : "memory");
}
__device__ __forceinline__ void wait_asynccnt0() {
  asm volatile("s_wait_asynccnt 0" ::: "memory");
}

// ---------------- elementwise helpers ----------------
__global__ void __launch_bounds__(256)
convert_kernel(const float* __restrict__ s, bf16_t* __restrict__ d, size_t n) {
  size_t i = (size_t)blockIdx.x * blockDim.x + threadIdx.x;
  if (i < n) d[i] = (bf16_t)s[i];
}

// (NL,Cout,Cin,KS) f32 -> (NL,KS,Cout,Cin) bf16 so each tap is row-contiguous
__global__ void __launch_bounds__(256)
packw_kernel(const float* __restrict__ w, bf16_t* __restrict__ wp,
             int Cout, int Cin, size_t n) {
  size_t i = (size_t)blockIdx.x * blockDim.x + threadIdx.x;
  if (i >= n) return;
  int k = (int)(i % KS);
  size_t r = i / KS;
  int ci = (int)(r % Cin); r /= Cin;
  int co = (int)(r % Cout);
  int l  = (int)(r / Cout);
  wp[(((size_t)l * KS + k) * Cout + co) * Cin + ci] = (bf16_t)w[i];
}

__global__ void __launch_bounds__(256)
embed_kernel(const int* __restrict__ tokens, const int* __restrict__ lens,
             const float* __restrict__ emb, float* __restrict__ x,
             bf16_t* __restrict__ xb) {
  size_t i = (size_t)blockIdx.x * blockDim.x + threadIdx.x;
  if (i >= (size_t)M * C) return;
  int c = (int)(i % C);
  size_t bt = i / C;
  int t = (int)(bt % T);
  int b = (int)(bt / T);
  int tok = tokens[bt];
  float mskf = (t < lens[b]) ? 1.f : 0.f;
  float v = emb[(size_t)tok * C + c] * 13.856406460551018f * mskf; // sqrt(192)
  x[i] = v;
  xb[i] = (bf16_t)v;
}

// qkv:(B,T,3C) -> vT:(B,NH,DH,T) so the P·V A-operand is contiguous over t
__global__ void __launch_bounds__(256)
vtrans_kernel(const bf16_t* __restrict__ qkv, bf16_t* __restrict__ vT) {
  int i = blockIdx.x * blockDim.x + threadIdx.x;
  if (i >= M * C) return;
  int t = i % T;
  int r = i / T;
  int d = r % DH;
  int bh = r / DH;
  int h = bh % NH;
  int b = bh / NH;
  vT[i] = qkv[((size_t)b * T + t) * SC + 2 * C + h * DH + d];
}

// ---------------- generic GEMM: out[m,n] = A[m,:]·W[n,:] + bias[n] ----------------
// wave computes 32x32 (2x2 WMMA tiles); block = 8 waves -> 128x64 tile
__global__ void __launch_bounds__(256)
gemm_bf16_kernel(const bf16_t* __restrict__ A, int lda,
                 const bf16_t* __restrict__ Bw, int ldb,
                 const float* __restrict__ bias,
                 float* __restrict__ Cf, bf16_t* __restrict__ Cb, int ldc,
                 int K, int relu) {
  const int lane = threadIdx.x & 31;
  const int wid  = threadIdx.x >> 5;
  const int row0 = blockIdx.y * 128 + (wid >> 1) * 32;
  const int col0 = blockIdx.x * 64 + (wid & 1) * 32;
  const int mrow = lane & 15;
  const int koff = (lane >> 4) * 8;
  const bf16_t* a0 = A  + (size_t)(row0 + mrow) * lda + koff;
  const bf16_t* a1 = a0 + (size_t)16 * lda;
  const bf16_t* b0 = Bw + (size_t)(col0 + mrow) * ldb + koff;
  const bf16_t* b1 = b0 + (size_t)16 * ldb;
  v8f acc[2][2] = {};
  for (int k0 = 0; k0 < K; k0 += 32) {
    v16bf va0 = load_row16(a0 + k0);
    v16bf va1 = load_row16(a1 + k0);
    v16bf vb0 = load_row16(b0 + k0);
    v16bf vb1 = load_row16(b1 + k0);
    acc[0][0] = wmma_bf16(va0, vb0, acc[0][0]);
    acc[0][1] = wmma_bf16(va0, vb1, acc[0][1]);
    acc[1][0] = wmma_bf16(va1, vb0, acc[1][0]);
    acc[1][1] = wmma_bf16(va1, vb1, acc[1][1]);
  }
  const int n  = lane & 15;
  const int mb = (lane >> 4) * 8;
#pragma unroll
  for (int i = 0; i < 2; ++i) {
#pragma unroll
    for (int j = 0; j < 2; ++j) {
      float bc = bias ? bias[col0 + j * 16 + n] : 0.f;
#pragma unroll
      for (int r = 0; r < 8; ++r) {
        float v = acc[i][j][r] + bc;
        if (relu) v = fmaxf(v, 0.f);
        size_t idx = (size_t)(row0 + i * 16 + mb + r) * ldc + (col0 + j * 16 + n);
        if (Cf) Cf[idx] = v;
        if (Cb) Cb[idx] = (bf16_t)v;
      }
    }
  }
}

// ---------------- conv1d (k=3, pad=1) as 3 shifted WMMA GEMMs ----------------
// X:(B,T,Cin) bf16, Wp:(KS,Cout,Cin) bf16 (this layer), out:(B,T,Cout)
// wave: 16 rows x 32 cols; block = 8 waves -> 64 rows x 64 cols
__global__ void __launch_bounds__(256)
conv1d_bf16_kernel(const bf16_t* __restrict__ X, const bf16_t* __restrict__ Wp,
                   const float* __restrict__ bias, const int* __restrict__ lens,
                   float* __restrict__ Yf, bf16_t* __restrict__ Yb,
                   int Cin, int Cout, int relu) {
  const int lane = threadIdx.x & 31;
  const int wid  = threadIdx.x >> 5;
  const int b    = blockIdx.z;
  const int row0 = blockIdx.y * 64 + (wid >> 1) * 16;
  const int col0 = blockIdx.x * 64 + (wid & 1) * 32;
  const int len  = lens[b];
  const int mrow = lane & 15;
  const int koff = (lane >> 4) * 8;
  v8f acc0 = {}, acc1 = {};
  for (int k = 0; k < KS; ++k) {
    int tr = row0 + mrow + k - 1;
    bool ok = (tr >= 0) && (tr < T) && (tr < len);
    const bf16_t* xrow = X + ((size_t)b * T + (size_t)(ok ? tr : 0)) * Cin + koff;
    const bf16_t* br0 = Wp + ((size_t)k * Cout + col0 + mrow) * Cin + koff;
    const bf16_t* br1 = br0 + (size_t)16 * Cin;
    for (int c0 = 0; c0 < Cin; c0 += 32) {
      v16bf a = {};
      if (ok) a = load_row16(xrow + c0);
      v16bf w0 = load_row16(br0 + c0);
      v16bf w1 = load_row16(br1 + c0);
      acc0 = wmma_bf16(a, w0, acc0);
      acc1 = wmma_bf16(a, w1, acc1);
    }
  }
  const int n  = lane & 15;
  const int mb = (lane >> 4) * 8;
#pragma unroll
  for (int j = 0; j < 2; ++j) {
    const v8f& acc = j ? acc1 : acc0;
    float bc = bias[col0 + j * 16 + n];
#pragma unroll
    for (int r = 0; r < 8; ++r) {
      int t = row0 + mb + r;
      float mskf = (t < len) ? 1.f : 0.f;
      float v = acc[r] + bc;
      if (relu) v = fmaxf(v, 0.f);
      v *= mskf;
      size_t idx = ((size_t)b * T + t) * Cout + (col0 + j * 16 + n);
      if (Yf) Yf[idx] = v;
      if (Yb) Yb[idx] = (bf16_t)v;
    }
  }
}

// ---------------- banded rel_v add helper (relv in LDS) ----------------
__device__ __forceinline__ void band_add(v8f (&oacc)[6], const bf16_t* relv,
                                         float p, int d, int shalf) {
  if (d < -RW || d > RW) return;
  const bf16_t* rv = relv + (size_t)(d + RW) * DH + shalf * 8;
#pragma unroll
  for (int dt = 0; dt < 6; ++dt)
#pragma unroll
    for (int r2 = 0; r2 < 8; ++r2)
      oacc[dt][r2] += p * (float)rv[dt * 16 + r2];
}

// ---------------- fused relative attention ----------------
// qkv:(B,T,3C) bf16, vT:(B,NH,DH,T) bf16, relk/relv:(9,DH) bf16 (this layer)
// grid: (T/64, NH, B), block: 128 (4 waves, one 16-row q tile each)
__global__ void __launch_bounds__(128)
attn_kernel(const bf16_t* __restrict__ qkv, const bf16_t* __restrict__ vT,
            const bf16_t* __restrict__ relk, const bf16_t* __restrict__ relv,
            const int* __restrict__ lens, bf16_t* __restrict__ obuf) {
  __shared__ float rdot[4][16][2 * RW + 1];
  __shared__ __align__(16) bf16_t s_relk[NREL];
  __shared__ __align__(16) bf16_t s_relv[NREL];
  const int lane = threadIdx.x & 31;
  const int w    = threadIdx.x >> 5;
  const int h    = blockIdx.y;
  const int b    = blockIdx.z;
  const int t0   = (blockIdx.x * 4 + w) * 16;
  const int len  = lens[b];
  const float inv = 0.10206207261596575f; // DH^-0.5
  const size_t base = (size_t)b * T * SC + (size_t)h * DH;
  const bf16_t* vTb = vT + ((size_t)b * NH + h) * DH * T;

  // stage rel tables into LDS via async copies (ASYNCcnt)
  for (int u = threadIdx.x; u < 2 * NRELU16; u += 128) {
    const bf16_t* src = (u < NRELU16) ? (relk + u * 8) : (relv + (u - NRELU16) * 8);
    bf16_t* dst = (u < NRELU16) ? (s_relk + u * 8) : (s_relv + (u - NRELU16) * 8);
    async_copy_b128(dst, src);
  }
  wait_asynccnt0();
  __syncthreads();

  // Phase A: banded q·rel_k dots (scaled)
  for (int idx = lane; idx < 16 * (2 * RW + 1); idx += 32) {
    int tt = idx / (2 * RW + 1);
    int d  = idx % (2 * RW + 1);
    const bf16_t* qp = qkv + base + (size_t)(t0 + tt) * SC;
    const bf16_t* rp = s_relk + (size_t)d * DH;
    float s = 0.f;
    for (int c = 0; c < DH; ++c) s += (float)qp[c] * (float)rp[c];
    rdot[w][tt][d] = s * inv;
  }
  __syncthreads();

  const int mrow  = lane & 15;
  const int koff  = (lane >> 4) * 8;
  const int tl    = lane & 15;
  const int t_abs = t0 + tl;
  const int shalf = lane >> 4;

  // Q operand registers (same per-lane pattern for A and B roles)
  v16bf qa[3];
  {
    const bf16_t* qp = qkv + base + (size_t)(t0 + mrow) * SC + koff;
#pragma unroll
    for (int ck = 0; ck < 3; ++ck) qa[ck] = load_row16(qp + ck * 32);
  }

  // Pass 1: per-row (t) global score max (S^T tiles: rows=s, cols=t)
  float mx = -1e30f;
  for (int s0 = 0; s0 < T; s0 += 16) {
    v8f st = {};
    const bf16_t* kp = qkv + base + C + (size_t)(s0 + mrow) * SC + koff;
#pragma unroll
    for (int ck = 0; ck < 3; ++ck) st = wmma_bf16(load_row16(kp + ck * 32), qa[ck], st);
    float lm = -1e30f;
#pragma unroll
    for (int r = 0; r < 8; ++r) {
      int s_abs = s0 + shalf * 8 + r;
      float sc = st[r] * inv;
      int d = s_abs - t_abs;
      if (d >= -RW && d <= RW) sc += rdot[w][tl][d + RW];
      if (s_abs >= len || t_abs >= len) sc -= 10000.f;
      lm = fmaxf(lm, sc);
    }
    lm = fmaxf(lm, __shfl_xor(lm, 16, 32));
    mx = fmaxf(mx, lm);
  }

  // Pass 2: p = exp(s - mx); O^T += V^T · P^T ; band rel_v
  float lsum = 0.f;
  v8f oacc[6] = {};
  for (int s0 = 0; s0 < T; s0 += 32) {
    v8f st0 = {}, st1 = {};
    const bf16_t* kp0 = qkv + base + C + (size_t)(s0 + mrow) * SC + koff;
    const bf16_t* kp1 = kp0 + (size_t)16 * SC;
#pragma unroll
    for (int ck = 0; ck < 3; ++ck) {
      st0 = wmma_bf16(load_row16(kp0 + ck * 32), qa[ck], st0);
      st1 = wmma_bf16(load_row16(kp1 + ck * 32), qa[ck], st1);
    }
    v8f p0, p1;
#pragma unroll
    for (int r = 0; r < 8; ++r) {
      int sa0 = s0 + shalf * 8 + r;
      int sa1 = sa0 + 16;
      float sc0 = st0[r] * inv;
      float sc1 = st1[r] * inv;
      int d0 = sa0 - t_abs, d1 = sa1 - t_abs;
      if (d0 >= -RW && d0 <= RW) sc0 += rdot[w][tl][d0 + RW];
      if (d1 >= -RW && d1 <= RW) sc1 += rdot[w][tl][d1 + RW];
      if (sa0 >= len || t_abs >= len) sc0 -= 10000.f;
      if (sa1 >= len || t_abs >= len) sc1 -= 10000.f;
      p0[r] = __expf(sc0 - mx);
      p1[r] = __expf(sc1 - mx);
      lsum += p0[r] + p1[r];
    }
    // C-layout p registers are exactly the B operand for O^T = V^T · P^T
    v16bf pb;
#pragma unroll
    for (int i = 0; i < 8; ++i) {
      pb[i]     = (bf16_t)p0[i];
      pb[8 + i] = (bf16_t)p1[i];
    }
#pragma unroll
    for (int dt = 0; dt < 6; ++dt) {
      const bf16_t* vrow = vTb + (size_t)(dt * 16 + mrow) * T + koff;
      v16bf va = load_row16(vrow + s0);
      oacc[dt] = wmma_bf16(va, pb, oacc[dt]);
    }
    // banded rel_v: only when this s-chunk overlaps [t0-RW, t0+15+RW]
    if (s0 + 31 >= t0 - RW && s0 <= t0 + 15 + RW) {
#pragma unroll
      for (int r = 0; r < 8; ++r) {
        float pown0 = p0[r];
        float pown1 = p1[r];
        float poth0 = __shfl_xor(p0[r], 16, 32);
        float poth1 = __shfl_xor(p1[r], 16, 32);
        int sown = s0 + shalf * 8 + r;
        int soth = s0 + (shalf ^ 1) * 8 + r;
        band_add(oacc, s_relv, pown0, sown - t_abs, shalf);
        band_add(oacc, s_relv, pown1, sown + 16 - t_abs, shalf);
        band_add(oacc, s_relv, poth0, soth - t_abs, shalf);
        band_add(oacc, s_relv, poth1, soth + 16 - t_abs, shalf);
      }
    }
  }
  lsum += __shfl_xor(lsum, 16, 32);
  float rl = 1.f / lsum;
  bf16_t* op = obuf + ((size_t)b * T + t_abs) * C + (size_t)h * DH + shalf * 8;
#pragma unroll
  for (int dt = 0; dt < 6; ++dt)
#pragma unroll
    for (int r = 0; r < 8; ++r) op[dt * 16 + r] = (bf16_t)(oacc[dt][r] * rl);
}

// ---------------- residual + LayerNorm (one wave per row) ----------------
__global__ void __launch_bounds__(256)
ln_kernel(float* __restrict__ x, const float* __restrict__ y,
          const float* __restrict__ g, const float* __restrict__ bta,
          bf16_t* __restrict__ xb) {
  const int lane = threadIdx.x & 31;
  const int row  = blockIdx.x * 8 + (threadIdx.x >> 5);
  float v[6];
  float s = 0.f;
#pragma unroll
  for (int j = 0; j < 6; ++j) {
    int c = j * 32 + lane;
    v[j] = x[(size_t)row * C + c] + y[(size_t)row * C + c];
    s += v[j];
  }
#pragma unroll
  for (int o = 16; o >= 1; o >>= 1) s += __shfl_xor(s, o, 32);
  float mu = s * (1.f / C);
  float vs = 0.f;
#pragma unroll
  for (int j = 0; j < 6; ++j) {
    float dd = v[j] - mu;
    vs += dd * dd;
  }
#pragma unroll
  for (int o = 16; o >= 1; o >>= 1) vs += __shfl_xor(vs, o, 32);
  float rstd = rsqrtf(vs * (1.f / C) + 1e-5f);
#pragma unroll
  for (int j = 0; j < 6; ++j) {
    int c = j * 32 + lane;
    float ov = (v[j] - mu) * rstd * g[c] + bta[c];
    x[(size_t)row * C + c] = ov;
    xb[(size_t)row * C + c] = (bf16_t)ov;
  }
}

// ---------------- outputs ----------------
__global__ void __launch_bounds__(256)
out_x_kernel(const float* __restrict__ x, const int* __restrict__ lens,
             float* __restrict__ out) {
  int i = blockIdx.x * blockDim.x + threadIdx.x;
  if (i >= M * C) return;
  int c = i % C;
  int bt = i / C;
  int t = bt % T;
  int b = bt / T;
  float mskf = (t < lens[b]) ? 1.f : 0.f;
  out[((size_t)b * C + c) * T + t] = x[i] * mskf;
}

__global__ void __launch_bounds__(256)
out_stats_kernel(const float* __restrict__ stats, const int* __restrict__ lens,
                 float* __restrict__ outm, float* __restrict__ outl) {
  int i = blockIdx.x * blockDim.x + threadIdx.x;
  if (i >= M * 2 * OUT) return;
  int o = i % (2 * OUT);
  int bt = i / (2 * OUT);
  int t = bt % T;
  int b = bt / T;
  float mskf = (t < lens[b]) ? 1.f : 0.f;
  float v = stats[i] * mskf;
  if (o < OUT) outm[((size_t)b * OUT + o) * T + t] = v;
  else         outl[((size_t)b * OUT + (o - OUT)) * T + t] = v;
}

__global__ void __launch_bounds__(256)
out_mask_kernel(const int* __restrict__ lens, float* __restrict__ om) {
  int i = blockIdx.x * blockDim.x + threadIdx.x;
  if (i >= M) return;
  int t = i % T;
  int b = i / T;
  om[i] = (t < lens[b]) ? 1.f : 0.f;
}

// ---------------- host orchestration ----------------
extern "C" void kernel_launch(void* const* d_in, const int* in_sizes, int n_in,
                              void* d_out, int out_size, void* d_ws, size_t ws_size,
                              hipStream_t stream) {
  (void)in_sizes; (void)n_in; (void)out_size; (void)ws_size;
  const int*   tokens = (const int*)d_in[0];
  const int*   lens   = (const int*)d_in[1];
  const float* emb    = (const float*)d_in[2];
  const float* wqkv   = (const float*)d_in[3];
  const float* bqkv   = (const float*)d_in[4];
  const float* wo     = (const float*)d_in[5];
  const float* bo     = (const float*)d_in[6];
  const float* relk   = (const float*)d_in[7];
  const float* relv   = (const float*)d_in[8];
  const float* ln1g   = (const float*)d_in[9];
  const float* ln1b   = (const float*)d_in[10];
  const float* ln2g   = (const float*)d_in[11];
  const float* ln2b   = (const float*)d_in[12];
  const float* w1     = (const float*)d_in[13];
  const float* b1     = (const float*)d_in[14];
  const float* w2     = (const float*)d_in[15];
  const float* b2     = (const float*)d_in[16];
  const float* pw     = (const float*)d_in[17];
  const float* pb     = (const float*)d_in[18];
  float* out = (float*)d_out;

  char* wsp = (char*)d_ws;
  auto alloc = [&](size_t bytes) -> char* {
    char* p = wsp;
    wsp += (bytes + 255) & ~(size_t)255;
    return p;
  };
  float*  x_f    = (float*)alloc((size_t)M * C * 4);
  float*  y_f    = (float*)alloc((size_t)M * C * 4);
  bf16_t* x_b    = (bf16_t*)alloc((size_t)M * C * 2);
  bf16_t* qkv_b  = (bf16_t*)alloc((size_t)M * SC * 2);
  bf16_t* vT_b   = (bf16_t*)alloc((size_t)M * C * 2);
  bf16_t* o_b    = (bf16_t*)alloc((size_t)M * C * 2);
  bf16_t* h_b    = (bf16_t*)alloc((size_t)M * DFF * 2);   // aliased as stats f32
  float*  stats  = (float*)h_b;                           // M*384*4 == M*768*2
  bf16_t* wqkv_b = (bf16_t*)alloc((size_t)NL * 3 * C * C * 2);
  bf16_t* wo_b   = (bf16_t*)alloc((size_t)NL * C * C * 2);
  bf16_t* w1_b   = (bf16_t*)alloc((size_t)NL * KS * DFF * C * 2);  // packed (KS,Cout,Cin)
  bf16_t* w2_b   = (bf16_t*)alloc((size_t)NL * KS * C * DFF * 2);  // packed (KS,Cout,Cin)
  bf16_t* pw_b   = (bf16_t*)alloc((size_t)2 * OUT * C * 2);
  bf16_t* relk_b = (bf16_t*)alloc((size_t)NL * NREL * 2);
  bf16_t* relv_b = (bf16_t*)alloc((size_t)NL * NREL * 2);

  auto cvt = [&](const float* s, bf16_t* d, size_t n) {
    convert_kernel<<<(unsigned)((n + 255) / 256), 256, 0, stream>>>(s, d, n);
  };
  cvt(wqkv, wqkv_b, (size_t)NL * 3 * C * C);
  cvt(wo,   wo_b,   (size_t)NL * C * C);
  cvt(pw,   pw_b,   (size_t)2 * OUT * C);
  cvt(relk, relk_b, (size_t)NL * NREL);
  cvt(relv, relv_b, (size_t)NL * NREL);
  {
    size_t n1 = (size_t)NL * DFF * C * KS;
    packw_kernel<<<(unsigned)((n1 + 255) / 256), 256, 0, stream>>>(w1, w1_b, DFF, C, n1);
    size_t n2 = (size_t)NL * C * DFF * KS;
    packw_kernel<<<(unsigned)((n2 + 255) / 256), 256, 0, stream>>>(w2, w2_b, C, DFF, n2);
  }

  embed_kernel<<<(M * C + 255) / 256, 256, 0, stream>>>(tokens, lens, emb, x_f, x_b);

  for (int l = 0; l < NL; ++l) {
    // QKV projection -> qkv_b (bf16, row stride 3C)
    gemm_bf16_kernel<<<dim3(SC / 64, M / 128), 256, 0, stream>>>(
        x_b, C, wqkv_b + (size_t)l * 3 * C * C, C, bqkv + (size_t)l * 3 * C,
        nullptr, qkv_b, SC, C, 0);
    // transpose V for contiguous P·V operand loads
    vtrans_kernel<<<(M * C + 255) / 256, 256, 0, stream>>>(qkv_b, vT_b);
    // fused relative attention -> o_b
    attn_kernel<<<dim3(T / 64, NH, B), 128, 0, stream>>>(
        qkv_b, vT_b, relk_b + (size_t)l * NREL, relv_b + (size_t)l * NREL,
        lens, o_b);
    // output projection -> y_f
    gemm_bf16_kernel<<<dim3(C / 64, M / 128), 256, 0, stream>>>(
        o_b, C, wo_b + (size_t)l * C * C, C, bo + (size_t)l * C,
        y_f, nullptr, C, C, 0);
    // x = LN(x + y)
    ln_kernel<<<M / 8, 256, 0, stream>>>(x_f, y_f, ln1g + (size_t)l * C,
                                         ln1b + (size_t)l * C, x_b);
    // FFN conv1 (relu, masked) -> h_b
    conv1d_bf16_kernel<<<dim3(DFF / 64, T / 64, B), 256, 0, stream>>>(
        x_b, w1_b + (size_t)l * KS * DFF * C, b1 + (size_t)l * DFF, lens,
        nullptr, h_b, C, DFF, 1);
    // FFN conv2 (masked) -> y_f
    conv1d_bf16_kernel<<<dim3(C / 64, T / 64, B), 256, 0, stream>>>(
        h_b, w2_b + (size_t)l * KS * C * DFF, b2 + (size_t)l * C, lens,
        y_f, nullptr, DFF, C, 0);
    // x = LN(x + y)
    ln_kernel<<<M / 8, 256, 0, stream>>>(x_f, y_f, ln2g + (size_t)l * C,
                                         ln2b + (size_t)l * C, x_b);
  }

  // final projection -> stats (f32, M x 2*OUT)
  gemm_bf16_kernel<<<dim3(2 * OUT / 64, M / 128), 256, 0, stream>>>(
      x_b, C, pw_b, C, pb, stats, nullptr, 2 * OUT, C, 0);

  float* out_x    = out;
  float* out_m    = out + (size_t)B * C * T;
  float* out_logs = out_m + (size_t)B * OUT * T;
  float* out_mask = out_logs + (size_t)B * OUT * T;
  out_x_kernel<<<(M * C + 255) / 256, 256, 0, stream>>>(x_f, lens, out_x);
  out_stats_kernel<<<(M * 2 * OUT + 255) / 256, 256, 0, stream>>>(stats, lens,
                                                                  out_m, out_logs);
  out_mask_kernel<<<(M + 255) / 256, 256, 0, stream>>>(lens, out_mask);
}
